// MultiHeadSelfAttention_83751862272686
// MI455X (gfx1250) — matrix operand order repaired and checked by reading, not compiled, not run on hardware
//
#include <hip/hip_runtime.h>

// ---------------- types ----------------
typedef __attribute__((ext_vector_type(16))) __bf16 bf16x16;
typedef __attribute__((ext_vector_type(8)))  __bf16 bf16x8;
typedef __attribute__((ext_vector_type(4)))  __bf16 bf16x4;
typedef __attribute__((ext_vector_type(8)))  float  f32x8;
typedef __attribute__((ext_vector_type(4)))  float  f32x4;
typedef __attribute__((ext_vector_type(4)))  unsigned int u32x4;
typedef __attribute__((ext_vector_type(8)))  int    i32x8;
typedef __attribute__((ext_vector_type(4)))  int    i32x4;

#define B_    2
#define S_LEN 2048
#define D_    2048
#define NUM_H 16
#define HD_   128
#define SM_SCALE 0.08838834764831845f  // 1/sqrt(128)

#if defined(__gfx1250__) && __has_builtin(__builtin_amdgcn_tensor_load_to_lds) && \
    __has_builtin(__builtin_amdgcn_s_wait_tensorcnt)
#define USE_TDM 1
#else
#define USE_TDM 0
#endif

__device__ __forceinline__ f32x8 zero8() {
    f32x8 z = {0.f, 0.f, 0.f, 0.f, 0.f, 0.f, 0.f, 0.f};
    return z;
}

// A-operand fragment (16x32, M x K): lane holds row m = lane%16,
// K = klo+{0..7} (elems 0..7) and klo+16+{0..7} (elems 8..15), klo = (lane<16)?0:8.
__device__ __forceinline__ bf16x16 load_afrag(const __bf16* p, int klo) {
    union { bf16x16 v; bf16x8 h[2]; } f;
    f.h[0] = *(const bf16x8*)(p + klo);
    f.h[1] = *(const bf16x8*)(p + klo + 16);
    return f.v;
}

// B-operand fragment (32x16, K x N): lane holds column n = lane%16,
// K = kb0+{0..7} (elems 0..7) and kb0+16+{0..7} (elems 8..15), kb0 = (lane<16)?0:8.  p points at row n of B^T.
__device__ __forceinline__ bf16x16 load_bfrag(const __bf16* p, int kb0) {
    union { bf16x16 v; bf16x8 h[2]; } f;
    f.h[0] = *(const bf16x8*)(p + kb0);
    f.h[1] = *(const bf16x8*)(p + kb0 + 16);
    return f.v;
}

__device__ __forceinline__ f32x8 wmma_bf16(bf16x16 a, bf16x16 b, f32x8 c) {
    // (neg_a, A, neg_b, B, c_mod, C, reuse_a, reuse_b)
    return __builtin_amdgcn_wmma_f32_16x16x32_bf16(false, a, false, b, (short)0, c,
                                                   false, false);
}

__device__ __forceinline__ float red_max16(float v) {
    v = fmaxf(v, __shfl_xor(v, 1));
    v = fmaxf(v, __shfl_xor(v, 2));
    v = fmaxf(v, __shfl_xor(v, 4));
    v = fmaxf(v, __shfl_xor(v, 8));
    return v;
}
__device__ __forceinline__ float red_sum16(float v) {
    v += __shfl_xor(v, 1);
    v += __shfl_xor(v, 2);
    v += __shfl_xor(v, 4);
    v += __shfl_xor(v, 8);
    return v;
}

#if USE_TDM
// Issue a 2D TDM tile load: tile_d1 rows of tile_d0 bf16 elements,
// row stride stride0 (elements).  All args must be wave-uniform.
// D# bit layout per CDNA5 ISA.
__device__ __forceinline__ void tdm_load_2d(const __bf16* gptr, __bf16* lptr,
                                            unsigned tile_d0, unsigned tile_d1,
                                            unsigned long long stride0) {
    unsigned long long ga = (unsigned long long)(uintptr_t)gptr;
    unsigned la = (unsigned)(uintptr_t)lptr;  // low 32 bits = LDS byte offset
    u32x4 g0;
    g0[0] = 1u;                                   // count=1, user mode
    g0[1] = la;                                   // lds_addr
    g0[2] = (unsigned)ga;                         // global_addr[95:64]
    g0[3] = (unsigned)((ga >> 32) & 0x01FFFFFFu)  // global_addr[120:96]
            | (2u << 30);                         // type = 2 ("image")
    i32x8 g1;
    g1[0] = (int)(1u << 16);                      // data_size = 1 (2 bytes)
    g1[1] = (int)((tile_d0 & 0xFFFFu) << 16);     // tensor_dim0[15:0] @ [63:48]
    g1[2] = (int)((tile_d0 >> 16) | ((tile_d1 & 0xFFFFu) << 16));  // t_dim0 hi | t_dim1 lo
    g1[3] = (int)((tile_d1 >> 16) | (tile_d0 << 16));              // t_dim1 hi | tile_dim0
    g1[4] = (int)(tile_d1 & 0xFFFFu);             // tile_dim1; tile_dim2 = 0
    g1[5] = (int)(unsigned)(stride0 & 0xFFFFFFFFull);   // dim0_stride[31:0]
    g1[6] = (int)(unsigned)((stride0 >> 32) & 0xFFFFull); // dim0_stride[47:32]; dim1_stride=0
    g1[7] = 0;
    i32x4 z4 = {0, 0, 0, 0};
#if __clang_major__ >= 23
    i32x8 z8 = {0, 0, 0, 0, 0, 0, 0, 0};
    __builtin_amdgcn_tensor_load_to_lds(g0, g1, z4, z4, z8, 0);
#else
    __builtin_amdgcn_tensor_load_to_lds(g0, g1, z4, z4, 0);
#endif
}
#endif  // USE_TDM

// ---------------- f32 -> bf16 convert ----------------
__global__ void cvt_bf16_kernel(const float* __restrict__ src, __bf16* __restrict__ dst,
                                int n4) {
    int i = blockIdx.x * blockDim.x + threadIdx.x;
    if (i < n4) {
        f32x4 f = *(const f32x4*)(src + (size_t)i * 4);
        bf16x4 o;
        o[0] = (__bf16)f[0];
        o[1] = (__bf16)f[1];
        o[2] = (__bf16)f[2];
        o[3] = (__bf16)f[3];
        *(bf16x4*)(dst + (size_t)i * 4) = o;
    }
}

// ---------------- GEMM: C[M=4096,N=2048] = A(bf16) * W(bf16)^T ----------------
// grid (N/64, M/128), block 128 (4 waves); wave -> 32x64 tile.
// mode 0: store bf16 to [B,H,S,Hd]   (q / k)
// mode 2: store bf16 to [B,H,Hd,S]   (v transposed)
// mode 3: store f32  to [B,S,D]      (final output)
__global__ __launch_bounds__(128) void gemm_bf16_kernel(
    const __bf16* __restrict__ A, const __bf16* __restrict__ W,
    __bf16* __restrict__ OB, float* __restrict__ OF, int mode) {
    const int tid  = threadIdx.x;
    const int lane = tid & 31;
    const int wv   = tid >> 5;
    const int lm   = lane & 15;
    const int klo  = (lane & 16) ? 8 : 0;
    const int kb0  = (lane & 16) ? 8 : 0;

    const int m0 = blockIdx.y * 128 + wv * 32;
    const int n0 = blockIdx.x * 64;

    f32x8 acc[2][4];
#pragma unroll
    for (int i = 0; i < 2; ++i)
#pragma unroll
        for (int j = 0; j < 4; ++j) acc[i][j] = zero8();

    const __bf16* arow0 = A + (size_t)(m0 + lm) * D_;
    const __bf16* arow1 = A + (size_t)(m0 + 16 + lm) * D_;
    const __bf16* wrow[4];
#pragma unroll
    for (int j = 0; j < 4; ++j) wrow[j] = W + (size_t)(n0 + j * 16 + lm) * D_;

    for (int kc = 0; kc < D_; kc += 32) {
        if (kc + 256 < D_) {  // L1 prefetch (global_prefetch_b8) a few tiles ahead
            __builtin_prefetch(arow0 + kc + 256, 0, 1);
            __builtin_prefetch(arow1 + kc + 256, 0, 1);
        }
        bf16x16 a0 = load_afrag(arow0 + kc, klo);
        bf16x16 a1 = load_afrag(arow1 + kc, klo);
#pragma unroll
        for (int j = 0; j < 4; ++j) {
            bf16x16 bb = load_bfrag(wrow[j] + kc, kb0);
            acc[0][j] = wmma_bf16(a0, bb, acc[0][j]);
            acc[1][j] = wmma_bf16(a1, bb, acc[1][j]);
        }
    }

    const int mb = (lane >> 4) * 8;  // C layout: VGPR r <-> row mb + r
#pragma unroll
    for (int i = 0; i < 2; ++i) {
#pragma unroll
        for (int r = 0; r < 8; ++r) {
            int gm = m0 + i * 16 + mb + r;
            int bi = gm >> 11;           // / S_LEN
            int s  = gm & (S_LEN - 1);
#pragma unroll
            for (int j = 0; j < 4; ++j) {
                int   gn  = n0 + j * 16 + lm;
                float val = acc[i][j][r];
                if (mode == 3) {
                    OF[(size_t)gm * D_ + gn] = val;
                } else {
                    int hh = gn >> 7;    // / HD_
                    int hd = gn & (HD_ - 1);
                    if (mode == 2)
                        OB[(((size_t)(bi * NUM_H + hh)) * HD_ + hd) * S_LEN + s] =
                            (__bf16)val;
                    else
                        OB[(((size_t)(bi * NUM_H + hh)) * S_LEN + s) * HD_ + hd] =
                            (__bf16)val;
                }
            }
        }
    }
}

// ---------------- flash attention ----------------
// grid (S/16, H, B), 1 wave per block. 16 queries x full Hd=128 per wave.
// K/V tiles staged in LDS via TDM (double-buffered), gated by TENSORcnt.
__global__ __launch_bounds__(32) void attn_kernel(const __bf16* __restrict__ Q,
                                                  const __bf16* __restrict__ K,
                                                  const __bf16* __restrict__ VT,
                                                  __bf16* __restrict__ AO) {
#if USE_TDM
    __shared__ __align__(16) __bf16 kt[2][32 * HD_];  // 32 kv rows x 128 (8 KB each)
    __shared__ __align__(16) __bf16 vt[2][HD_ * 32];  // 128 d rows x 32  (8 KB each)
#endif
    __shared__ __align__(16) __bf16 pbuf[16 * 32];    // P tile, row-major 16x32

    const int lane = threadIdx.x;
    const int lm   = lane & 15;
    const int klo  = (lane & 16) ? 8 : 0;
    const int kb0  = (lane & 16) ? 8 : 0;
    const int qt = blockIdx.x, h = blockIdx.y, b = blockIdx.z;

    const size_t head_off = (size_t)(b * NUM_H + h) * (size_t)S_LEN * HD_;
    const __bf16* qbase = Q + head_off + (size_t)(qt * 16) * HD_;
    const __bf16* kbase = K + head_off;
    const __bf16* vbase = VT + head_off;  // [HD][S] per head

    bf16x16 qa[4];
#pragma unroll
    for (int c = 0; c < 4; ++c) qa[c] = load_afrag(qbase + lm * HD_ + c * 32, klo);

    f32x8 acc[8];
#pragma unroll
    for (int d = 0; d < 8; ++d) acc[d] = zero8();
    float mrow[8], lrow[8];
#pragma unroll
    for (int r = 0; r < 8; ++r) { mrow[r] = -1e30f; lrow[r] = 0.f; }

    const int mb = (lane >> 4) * 8;

#if USE_TDM
    tdm_load_2d(kbase, kt[0], HD_, 32, HD_);       // K rows jb..jb+31
    tdm_load_2d(vbase, vt[0], 32, HD_, S_LEN);     // V^T rows 0..127, cols jb..jb+31
#endif

    int cur = 0;
    for (int jb = 0; jb < S_LEN; jb += 32, cur ^= 1) {
#if USE_TDM
        if (jb + 32 < S_LEN) {
            tdm_load_2d(kbase + (size_t)(jb + 32) * HD_, kt[cur ^ 1], HD_, 32, HD_);
            tdm_load_2d(vbase + (jb + 32), vt[cur ^ 1], 32, HD_, S_LEN);
            __builtin_amdgcn_s_wait_tensorcnt((short)2);  // current buffers ready
        } else {
            __builtin_amdgcn_s_wait_tensorcnt((short)0);
        }
        asm volatile("" ::: "memory");
        const __bf16* ktile = kt[cur];
        const __bf16* vtile = vt[cur];
#endif
        // ---- scores: two 16x16 tiles over K=128 in chunks of 32 ----
        f32x8 s0 = zero8(), s1 = zero8();
#pragma unroll
        for (int c = 0; c < 4; ++c) {
#if USE_TDM
            bf16x16 k0 = load_bfrag(ktile + lm * HD_ + c * 32, kb0);
            bf16x16 k1 = load_bfrag(ktile + (16 + lm) * HD_ + c * 32, kb0);
#else
            bf16x16 k0 = load_bfrag(kbase + (size_t)(jb + lm) * HD_ + c * 32, kb0);
            bf16x16 k1 = load_bfrag(kbase + (size_t)(jb + 16 + lm) * HD_ + c * 32, kb0);
#endif
            s0 = wmma_bf16(qa[c], k0, s0);
            s1 = wmma_bf16(qa[c], k1, s1);
        }
        // ---- online softmax (row = mb + r lives across the lane's 16-half) ----
        float p0v[8], p1v[8];
#pragma unroll
        for (int r = 0; r < 8; ++r) {
            float a0   = s0[r] * SM_SCALE;
            float a1   = s1[r] * SM_SCALE;
            float mx   = red_max16(fmaxf(a0, a1));
            float mn   = fmaxf(mrow[r], mx);
            float corr = __expf(mrow[r] - mn);
            float p0   = __expf(a0 - mn);
            float p1   = __expf(a1 - mn);
            float rs   = red_sum16(p0 + p1);
            lrow[r] = lrow[r] * corr + rs;
            mrow[r] = mn;
#pragma unroll
            for (int d = 0; d < 8; ++d) acc[d][r] *= corr;
            p0v[r] = p0;
            p1v[r] = p1;
        }
        // ---- transpose P (C layout -> A layout) through LDS ----
#pragma unroll
        for (int r = 0; r < 8; ++r) {
            pbuf[(mb + r) * 32 + lm]      = (__bf16)p0v[r];
            pbuf[(mb + r) * 32 + 16 + lm] = (__bf16)p1v[r];
        }
        __syncthreads();
        union { bf16x16 v; bf16x8 h[2]; } pf;
        pf.h[0] = *(const bf16x8*)(pbuf + lm * 32 + klo);
        pf.h[1] = *(const bf16x8*)(pbuf + lm * 32 + klo + 16);
        __syncthreads();
        // ---- acc += P (16x32) * V (32x128) ----
#pragma unroll
        for (int d = 0; d < 8; ++d) {
#if USE_TDM
            bf16x16 vf = load_bfrag(vtile + (d * 16 + lm) * 32, kb0);
#else
            bf16x16 vf = load_bfrag(vbase + (size_t)(d * 16 + lm) * S_LEN + jb, kb0);
#endif
            acc[d] = wmma_bf16(pf.v, vf, acc[d]);
        }
    }

    // ---- epilogue: normalize, store to [B,S,D] bf16 (heads interleaved) ----
#pragma unroll
    for (int r = 0; r < 8; ++r) {
        float inv  = 1.0f / lrow[r];
        int   srow = qt * 16 + mb + r;
        size_t rowoff = ((size_t)b * S_LEN + srow) * D_ + h * HD_;
#pragma unroll
        for (int d = 0; d < 8; ++d)
            AO[rowoff + d * 16 + lm] = (__bf16)(acc[d][r] * inv);
    }
}

// ---------------- launcher ----------------
extern "C" void kernel_launch(void* const* d_in, const int* in_sizes, int n_in,
                              void* d_out, int out_size, void* d_ws, size_t ws_size,
                              hipStream_t stream) {
    (void)in_sizes; (void)n_in; (void)out_size; (void)ws_size;
    const float* x  = (const float*)d_in[0];
    const float* Wq = (const float*)d_in[1];
    const float* Wk = (const float*)d_in[2];
    const float* Wv = (const float*)d_in[3];
    const float* Wo = (const float*)d_in[4];

    const size_t XN = (size_t)B_ * S_LEN * D_;  // 8,388,608
    const size_t WN = (size_t)D_ * D_;          // 4,194,304

    char*   ws   = (char*)d_ws;
    size_t  off  = 0;
    __bf16* xbf  = (__bf16*)(ws + off); off += XN * 2;
    __bf16* wqbf = (__bf16*)(ws + off); off += WN * 2;
    __bf16* wkbf = (__bf16*)(ws + off); off += WN * 2;
    __bf16* wvbf = (__bf16*)(ws + off); off += WN * 2;
    __bf16* wobf = (__bf16*)(ws + off); off += WN * 2;
    __bf16* qbf  = (__bf16*)(ws + off); off += XN * 2;
    __bf16* kbf  = (__bf16*)(ws + off); off += XN * 2;
    __bf16* vtbf = (__bf16*)(ws + off); off += XN * 2;
    __bf16* aobf = (__bf16*)(ws + off); off += XN * 2;

    // f32 -> bf16
    cvt_bf16_kernel<<<(int)(XN / 4 / 256), 256, 0, stream>>>(x, xbf, (int)(XN / 4));
    cvt_bf16_kernel<<<(int)(WN / 4 / 256), 256, 0, stream>>>(Wq, wqbf, (int)(WN / 4));
    cvt_bf16_kernel<<<(int)(WN / 4 / 256), 256, 0, stream>>>(Wk, wkbf, (int)(WN / 4));
    cvt_bf16_kernel<<<(int)(WN / 4 / 256), 256, 0, stream>>>(Wv, wvbf, (int)(WN / 4));
    cvt_bf16_kernel<<<(int)(WN / 4 / 256), 256, 0, stream>>>(Wo, wobf, (int)(WN / 4));

    dim3 ggrid(D_ / 64, (B_ * S_LEN) / 128);
    gemm_bf16_kernel<<<ggrid, 128, 0, stream>>>(xbf, wqbf, qbf, nullptr, 0);
    gemm_bf16_kernel<<<ggrid, 128, 0, stream>>>(xbf, wkbf, kbf, nullptr, 0);
    gemm_bf16_kernel<<<ggrid, 128, 0, stream>>>(xbf, wvbf, vtbf, nullptr, 2);

    dim3 agrid(S_LEN / 16, NUM_H, B_);
    attn_kernel<<<agrid, 32, 0, stream>>>(qbf, kbf, vtbf, aobf);

    gemm_bf16_kernel<<<ggrid, 128, 0, stream>>>(aobf, wobf, nullptr, (float*)d_out, 3);
}
